// QKVAttention_32701880991893
// MI455X (gfx1250) — matrix operand order repaired and checked by reading, not compiled, not run on hardware
//
#include <hip/hip_runtime.h>
#include <hip/hip_bf16.h>
#include <stdint.h>

typedef __attribute__((ext_vector_type(16))) __bf16 v16bf;
typedef __attribute__((ext_vector_type(8)))  __bf16 v8bf;
typedef __attribute__((ext_vector_type(8)))  float  v8f;

#define T_LEN 2048
#define NH    16
#define CH    64
#define SBLK  64
#define CSTR  72    // bf16 stride for K/V LDS tiles (144 B rows, 16B aligned)
#define ROWP  24    // bf16 stride for P LDS tile   (48 B cols, 16B aligned)
#define NEG_BIG (-3.4028235e38f)

static __device__ __forceinline__ __bf16 to_bf16(float x) { return (__bf16)x; }

union V16U { v16bf v; v8bf h[2]; };

__global__ __launch_bounds__(128) void attn_fused_32701880991893(
    const float* __restrict__ qkv,
    const uint8_t* __restrict__ mask,
    float* __restrict__ out)
{
  const int tid  = threadIdx.x;
  const int wave = tid >> 5;
  const int lane = tid & 31;
  const int col  = lane & 15;
  const int hi   = lane >> 4;

  const int bh   = blockIdx.x & 31;
  const int qblk = blockIdx.x >> 5;
  const int b = bh >> 4, h = bh & 15;
  const int t0 = qblk * 64 + wave * 16;

  const size_t bo = (size_t)b * (3 * NH * CH) * T_LEN;
  const float* qp = qkv + bo + (size_t)(h * CH) * T_LEN;
  const float* kp = qkv + bo + (size_t)((NH + h) * CH) * T_LEN;
  const float* vp = qkv + bo + (size_t)((2 * NH + h) * CH) * T_LEN;
  const uint8_t* mp = mask + (size_t)b * T_LEN;

  __shared__ __align__(16) __bf16 kbuf[SBLK * CSTR];   // K transposed [s][c]
  __shared__ __align__(16) __bf16 vbuf[CH * CSTR];     // V [c][s]
  __shared__ __align__(16) __bf16 ldsP[4][SBLK * ROWP];
  __bf16* pbuf = &ldsP[wave][0];

  const int crow  = tid >> 1;
  const int shalf = (tid & 1) * 32;

  // ---- Q A-fragments (16t x 64c), scale^2 = 1/8 folded in
  v16bf qa[2];
  const int trow = t0 + col;
  #pragma unroll
  for (int kc = 0; kc < 2; ++kc) {
    #pragma unroll
    for (int e = 0; e < 16; ++e) {
      int k = e + 16 * hi;
      qa[kc][e] = to_bf16(qp[(size_t)(kc * 32 + k) * T_LEN + trow] * 0.125f);
    }
  }

  bool qinv[8];
  #pragma unroll
  for (int j = 0; j < 8; ++j) qinv[j] = (mp[t0 + j + 8 * hi] == 0);

  float mrow[8], lrow[8];
  #pragma unroll
  for (int j = 0; j < 8; ++j) { mrow[j] = -INFINITY; lrow[j] = 0.0f; }
  v8f oacc[4] = {};

  for (int s0 = 0; s0 < T_LEN; s0 += SBLK) {
    // ---- key-mask bytes: issue first so they're in flight during staging
    bool kinv[4];
    #pragma unroll
    for (int nt = 0; nt < 4; ++nt) kinv[nt] = (mp[s0 + nt * 16 + col] == 0);

    // ---- cooperative staging: K -> kbuf[s][c] (transposed), V -> vbuf[c][s]
    {
      const float4* ksrc = (const float4*)(kp + (size_t)crow * T_LEN + s0 + shalf);
      const float4* vsrc = (const float4*)(vp + (size_t)crow * T_LEN + s0 + shalf);
      if (s0 + SBLK < T_LEN) {         // global_prefetch_b8 for next block
        __builtin_prefetch(kp + (size_t)crow * T_LEN + s0 + SBLK + shalf, 0, 3);
        __builtin_prefetch(vp + (size_t)crow * T_LEN + s0 + SBLK + shalf, 0, 3);
      }
      #pragma unroll
      for (int q4 = 0; q4 < 8; ++q4) {  // K: scattered bf16 stores (transpose)
        float4 f = ksrc[q4];
        int sl = shalf + q4 * 4;
        kbuf[(sl + 0) * CSTR + crow] = to_bf16(f.x);
        kbuf[(sl + 1) * CSTR + crow] = to_bf16(f.y);
        kbuf[(sl + 2) * CSTR + crow] = to_bf16(f.z);
        kbuf[(sl + 3) * CSTR + crow] = to_bf16(f.w);
      }
      #pragma unroll
      for (int q4 = 0; q4 < 8; q4 += 2) { // V: packed 16B ds_store_b128
        float4 f0 = vsrc[q4], f1 = vsrc[q4 + 1];
        v8bf pk;
        pk[0] = to_bf16(f0.x); pk[1] = to_bf16(f0.y);
        pk[2] = to_bf16(f0.z); pk[3] = to_bf16(f0.w);
        pk[4] = to_bf16(f1.x); pk[5] = to_bf16(f1.y);
        pk[6] = to_bf16(f1.z); pk[7] = to_bf16(f1.w);
        *(v8bf*)(&vbuf[crow * CSTR + shalf + q4 * 4]) = pk;
      }
    }
    __syncthreads();

    // ---- GEMM1: preload ALL 8 K B-fragments, then 8 back-to-back WMMAs
    V16U kb[8];
    #pragma unroll
    for (int nt = 0; nt < 4; ++nt) {
      const int base = (nt * 16 + col) * CSTR + 16 * hi;
      kb[nt].h[0]     = *(const v8bf*)(&kbuf[base]);
      kb[nt].h[1]     = *(const v8bf*)(&kbuf[base + 8]);
      kb[nt + 4].h[0] = *(const v8bf*)(&kbuf[base + 32]);
      kb[nt + 4].h[1] = *(const v8bf*)(&kbuf[base + 40]);
    }
    v8f sacc[4];
    #pragma unroll
    for (int nt = 0; nt < 4; ++nt) {
      v8f acc = {};
      acc = __builtin_amdgcn_wmma_f32_16x16x32_bf16(false, qa[0], false, kb[nt].v,
                                                    (short)0, acc, false, false);
      acc = __builtin_amdgcn_wmma_f32_16x16x32_bf16(false, qa[1], false, kb[nt + 4].v,
                                                    (short)0, acc, false, false);
      sacc[nt] = acc;
    }

    // ---- mask + online softmax over these 64 columns
    float pval[4][8];
    #pragma unroll
    for (int j = 0; j < 8; ++j) {
      float mloc = NEG_BIG;
      #pragma unroll
      for (int nt = 0; nt < 4; ++nt) {
        float sv = (kinv[nt] || qinv[j]) ? NEG_BIG : sacc[nt][j];
        pval[nt][j] = sv;
        mloc = fmaxf(mloc, sv);
      }
      #pragma unroll
      for (int off = 1; off < 16; off <<= 1)
        mloc = fmaxf(mloc, __shfl_xor(mloc, off, 32));
      float newm = fmaxf(mrow[j], mloc);
      float alpha = __expf(mrow[j] - newm);
      mrow[j] = newm;
      float ssum = 0.0f;
      #pragma unroll
      for (int nt = 0; nt < 4; ++nt) {
        float p = __expf(pval[nt][j] - newm);
        pval[nt][j] = p;
        ssum += p;
      }
      #pragma unroll
      for (int off = 1; off < 16; off <<= 1)
        ssum += __shfl_xor(ssum, off, 32);
      lrow[j] = lrow[j] * alpha + ssum;
      #pragma unroll
      for (int ct = 0; ct < 4; ++ct) oacc[ct][j] *= alpha;
    }

    // ---- stage P (col-major): one packed 16B store per N-tile per lane
    #pragma unroll
    for (int nt = 0; nt < 4; ++nt) {
      v8bf pk;
      #pragma unroll
      for (int j = 0; j < 8; ++j) pk[j] = to_bf16(pval[nt][j]);
      *(v8bf*)(&pbuf[(nt * 16 + col) * ROWP + hi * 8]) = pk;
    }
    __syncthreads();

    // ---- GEMM2: per K-chunk, preload pa + all 4 V fragments, then 4 WMMAs
    #pragma unroll
    for (int kc = 0; kc < 2; ++kc) {
      v16bf pa;
      #pragma unroll
      for (int e = 0; e < 16; ++e) {
        int k = e + 16 * hi;
        pa[e] = pbuf[(kc * 32 + k) * ROWP + col];
      }
      V16U vb[4];
      #pragma unroll
      for (int ct = 0; ct < 4; ++ct) {
        const int base = (ct * 16 + col) * CSTR + kc * 32 + 16 * hi;
        vb[ct].h[0] = *(const v8bf*)(&vbuf[base]);
        vb[ct].h[1] = *(const v8bf*)(&vbuf[base + 8]);
      }
      #pragma unroll
      for (int ct = 0; ct < 4; ++ct)
        oacc[ct] = __builtin_amdgcn_wmma_f32_16x16x32_bf16(false, pa, false, vb[ct].v,
                                                           (short)0, oacc[ct],
                                                           false, false);
    }
    __syncthreads();   // protect kbuf/vbuf/pbuf before next restage
  }

  // ---- normalize (reciprocal once per row) and write O: out[b][h*CH + c][t]
  float rinv[8];
  #pragma unroll
  for (int j = 0; j < 8; ++j) rinv[j] = 1.0f / lrow[j];
  float* op = out + (size_t)b * (NH * CH) * T_LEN + (size_t)h * CH * T_LEN;
  #pragma unroll
  for (int ct = 0; ct < 4; ++ct) {
    int c = ct * 16 + col;
    #pragma unroll
    for (int j = 0; j < 8; ++j) {
      int t = t0 + 8 * hi + j;
      op[(size_t)c * T_LEN + t] = oacc[ct][j] * rinv[j];
    }
  }
}

extern "C" void kernel_launch(void* const* d_in, const int* in_sizes, int n_in,
                              void* d_out, int out_size, void* d_ws, size_t ws_size,
                              hipStream_t stream) {
  const float*   qkv = (const float*)d_in[0];
  const uint8_t* msk = (const uint8_t*)d_in[1];   // jnp.bool_ -> 1 byte
  float* out = (float*)d_out;
  (void)in_sizes; (void)n_in; (void)out_size; (void)d_ws; (void)ws_size;
  dim3 grid(32 * (T_LEN / 64));                   // 32 bh * 32 q-blocks
  attn_fused_32701880991893<<<grid, 128, 0, stream>>>(qkv, msk, out);
}
